// lianglu_att_13864154431710
// MI455X (gfx1250) — compile-verified, matrix operand order checked by reading, not run on hardware
//
#include <hip/hip_runtime.h>
#include <hip/hip_bf16.h>

typedef _Float16 half_t;
typedef __attribute__((ext_vector_type(16))) _Float16 v16h;
typedef __attribute__((ext_vector_type(8)))  _Float16 v8h;
typedef __attribute__((ext_vector_type(8)))  float    v8f;

// ---------------------------------------------------------------------------
// WMMA helpers (gfx1250: V_WMMA_F32_16X16X32_F16, wave32)
// ---------------------------------------------------------------------------
__device__ __forceinline__ v8f wmma32(v16h a, v16h b, v8f c) {
  return __builtin_amdgcn_wmma_f32_16x16x32_f16(false, a, false, b, (short)0, c,
                                                false, false);
}

// k-contiguous fragment: two b128 loads, no packing moves.
// Caller passes p offset to (lane's row/col base, k0 + 8*(lane>>4)).
// ISA 7.12.2: elements 0..7 -> k = kb+0..7 ; elements 8..15 -> k = kb+16..23.
__device__ __forceinline__ v16h frag_k1(const half_t* p) {
  union { v16h v; v8h h[2]; } u;
  u.h[0] = *(const v8h*)(p);
  u.h[1] = *(const v8h*)(p + 16);
  return u.v;
}

// k-strided fragment (only used for the tiny 32x32 ctx^T operand).
__device__ __forceinline__ v16h frag_ks(const half_t* p, long ks) {
  v16h r;
#pragma unroll
  for (int e = 0; e < 8; ++e) {
    r[e]     = p[(long)e * ks];
    r[8 + e] = p[(long)(16 + e) * ks];
  }
  return r;
}

// ---------------------------------------------------------------------------
// Generic batched WMMA GEMM:  Y[z] = A[z] (MxK) @ B[z] (KxN) (+bias)(+resid)
//   A : MxK row-major (AT: stored KxM, lda = row stride) -> frag_k1 / frag_ks
//   Bt: B^T, i.e. N rows x K cols row-major (k contiguous)  -> frag_k1
// Each wave computes an (MT*16) x 16 strip: one B fragment feeds MT WMMAs
// per k-step (4x less activation re-fetch; 2.5 b128 loads per wmma at MT=4).
// Outputs (any subset):
//   Yt: spatial-major f16 (N x M, row stride ldyt) -- per-lane v8h stores
//   Yc: channel-major f16 (M x N, row stride ldc)  -- strided stores
//   Fc: channel-major f32 (M x N, row stride ldc)  -- strided stores
// Batch offsets: z -> (z>>zlog)*hi + (z&((1<<zlog)-1))*lo per operand, which
// lets one kernel serve per-batch and per-(batch,head) tensor views.
// ---------------------------------------------------------------------------
template <int MT, bool AT>
__global__ void gemm_nt_kernel(const half_t* __restrict__ A, long a_hi, long a_lo, int lda,
                               const half_t* __restrict__ Bt, long b_hi, long b_lo, int ldb,
                               const float* __restrict__ bias,
                               const half_t* __restrict__ residT, long r_hi, long r_lo,
                               half_t* __restrict__ Yt, long yt_hi, long yt_lo, int ldyt,
                               half_t* __restrict__ Yc, long yc_hi, long yc_lo,
                               float* __restrict__ Fc, long fc_hi, long fc_lo,
                               int ldc, int zlog, int M, int N, int K) {
  const int wid  = threadIdx.x >> 5;
  const int lane = threadIdx.x & 31;
  const int hi   = lane >> 4;
  const int lo   = lane & 15;
  const int n0   = (blockIdx.x * (blockDim.x >> 5) + wid) * 16;
  if (n0 >= N) return;
  const int mbase = blockIdx.y * MT * 16;
  const long zb = blockIdx.z >> zlog;
  const long zh = blockIdx.z & ((1 << zlog) - 1);

  const half_t* Az = A  + zb * a_hi + zh * a_lo;
  const half_t* Bz = Bt + zb * b_hi + zh * b_lo;

  v8f acc[MT];
#pragma unroll
  for (int mt = 0; mt < MT; ++mt) acc[mt] = (v8f){};

  for (int k0 = 0; k0 < K; k0 += 32) {
    const v16h b = frag_k1(Bz + (long)(n0 + lo) * ldb + k0 + hi * 8);
    v16h a[MT];
#pragma unroll
    for (int mt = 0; mt < MT; ++mt) {
      const int m0 = mbase + mt * 16;
      if (AT) a[mt] = frag_ks(Az + (long)(k0 + hi * 8) * lda + m0 + lo, lda);
      else    a[mt] = frag_k1(Az + (long)(m0 + lo) * lda + k0 + hi * 8);
    }
#pragma unroll
    for (int mt = 0; mt < MT; ++mt) acc[mt] = wmma32(a[mt], b, acc[mt]);
  }

  // D tile mt: row m = mbase + mt*16 + r + 8*hi (contiguous over r), n = n0+lo.
  const int n = n0 + lo;
#pragma unroll
  for (int mt = 0; mt < MT; ++mt) {
    const int m0 = mbase + mt * 16;
    float vals[8];
#pragma unroll
    for (int r = 0; r < 8; ++r) vals[r] = acc[mt][r];
    if (bias) {
      const float* bp = bias + m0 + hi * 8;
#pragma unroll
      for (int r = 0; r < 8; ++r) vals[r] += bp[r];
    }
    if (residT) {
      const v8h rv = *(const v8h*)(residT + zb * r_hi + zh * r_lo +
                                   (long)n * ldyt + m0 + hi * 8);
#pragma unroll
      for (int r = 0; r < 8; ++r) vals[r] += (float)rv[r];
    }
    if (Yt) {
      v8h hv;
#pragma unroll
      for (int r = 0; r < 8; ++r) hv[r] = (half_t)vals[r];
      *(v8h*)(Yt + zb * yt_hi + zh * yt_lo + (long)n * ldyt + m0 + hi * 8) = hv;
    }
    if (Yc) {
      half_t* cp = Yc + zb * yc_hi + zh * yc_lo + (long)(m0 + hi * 8) * ldc + n;
#pragma unroll
      for (int r = 0; r < 8; ++r) cp[(long)r * ldc] = (half_t)vals[r];
    }
    if (Fc) {
      float* fp = Fc + zb * fc_hi + zh * fc_lo + (long)(m0 + hi * 8) * ldc + n;
#pragma unroll
      for (int r = 0; r < 8; ++r) fp[(long)r * ldc] = vals[r];
    }
  }
}

// ---------------------------------------------------------------------------
// fp32 -> fp16 conversion (grid-stride) for weights
// ---------------------------------------------------------------------------
__global__ void cvt_f32_f16_kernel(const float* __restrict__ src,
                                   half_t* __restrict__ dst, long n) {
  long i = (long)blockIdx.x * blockDim.x + threadIdx.x;
  const long stride = (long)gridDim.x * blockDim.x;
  for (; i < n; i += stride) dst[i] = (half_t)src[i];
}

// ---------------------------------------------------------------------------
// LDS-tiled transpose + convert: x (C x N f32, channel-major) -> xT (N x C f16)
// 32x32 tiles, coalesced on both sides. grid (N/32, C/32, B), block 256.
// ---------------------------------------------------------------------------
__global__ void transpose_cvt_kernel(const float* __restrict__ src,
                                     half_t* __restrict__ dst, int C, int N) {
  __shared__ float tile[32][33];
  const int b  = blockIdx.z;
  const int c0 = blockIdx.y * 32;
  const int n0 = blockIdx.x * 32;
  const int tx = threadIdx.x & 31;
  const int ty = threadIdx.x >> 5;  // 8 rows per pass
  const float* s = src + (long)b * C * N;
  half_t*      d = dst + (long)b * C * N;
#pragma unroll
  for (int r = ty; r < 32; r += 8)
    tile[r][tx] = s[(long)(c0 + r) * N + n0 + tx];
  __syncthreads();
#pragma unroll
  for (int r = ty; r < 32; r += 8)
    d[(long)(n0 + r) * C + c0 + tx] = (half_t)tile[tx][r];
}

// ---------------------------------------------------------------------------
// Offset network on spatial-major q: depthwise 6x6 stride 4 pad 1 (per group
// of 32 channels), tanh-GELU, 2x32 pointwise, tanh*4 -> full-res pixel coords.
// grid: 64 (b*g), block: 256 (16x16 downsampled positions).
// ---------------------------------------------------------------------------
__global__ void offsets_kernel(const half_t* __restrict__ qT,
                               const float* __restrict__ dw,   // (32,1,6,6)
                               const float* __restrict__ pw,   // (2,32)
                               float* __restrict__ px, float* __restrict__ py) {
  const int t  = threadIdx.x;
  const int bg = blockIdx.x;
  const int b  = bg >> 3, g = bg & 7;
  const int ys = t >> 4, xs = t & 15;
  const half_t* qbase = qT + (long)b * 1048576 + g * 32;

  float hreg[32];
#pragma unroll
  for (int c = 0; c < 32; ++c) hreg[c] = 0.f;

  for (int ky = 0; ky < 6; ++ky) {
    const int y = ys * 4 - 1 + ky;
    if (y < 0 || y >= 64) continue;
    for (int kx = 0; kx < 6; ++kx) {
      const int x = xs * 4 - 1 + kx;
      if (x < 0 || x >= 64) continue;
      const half_t* qp = qbase + (long)(y * 64 + x) * 256;  // 32 contiguous ch
      const int tap = ky * 6 + kx;
#pragma unroll
      for (int c = 0; c < 32; ++c)
        hreg[c] += dw[c * 36 + tap] * (float)qp[c];
    }
  }
  float o0 = 0.f, o1 = 0.f;
#pragma unroll
  for (int c = 0; c < 32; ++c) {
    const float a  = hreg[c];
    const float g3 = a + 0.044715f * a * a * a;             // jax.nn.gelu (tanh)
    const float ge = 0.5f * a * (1.f + tanhf(0.7978845608028654f * g3));
    o0 += pw[c] * ge;
    o1 += pw[32 + c] * ge;
  }
  const float vx = (float)xs + tanhf(o0) * 4.0f;
  const float vy = (float)ys + tanhf(o1) * 4.0f;
  const int idx = bg * 256 + t;
  px[idx] = vx * (64.0f / 15.0f) - 0.5f;   // ((2v/15-1)+1)*0.5*64 - 0.5
  py[idx] = vy * (64.0f / 15.0f) - 0.5f;
}

// ---------------------------------------------------------------------------
// Bilinear gather (zeros padding) from spatial-major feat; channels are
// contiguous so each corner is four v8h loads. kv written spatial-major.
// ---------------------------------------------------------------------------
__global__ void sample_kernel(const half_t* __restrict__ featT,
                              const float* __restrict__ px,
                              const float* __restrict__ py,
                              half_t* __restrict__ kvT) {
  const int t  = threadIdx.x;
  const int bg = blockIdx.x;
  const int b  = bg >> 3, g = bg & 7;
  const int idx = bg * 256 + t;
  const float fx = px[idx], fy = py[idx];
  const float x0f = floorf(fx), y0f = floorf(fy);
  const int x0 = (int)x0f, y0 = (int)y0f;
  const float wx1 = fx - x0f, wy1 = fy - y0f;
  const float wx0 = 1.f - wx1, wy0 = 1.f - wy1;
  const bool vx0 = (x0 >= 0) && (x0 < 64), vx1 = (x0 + 1 >= 0) && (x0 + 1 < 64);
  const bool vy0 = (y0 >= 0) && (y0 < 64), vy1 = (y0 + 1 >= 0) && (y0 + 1 < 64);
  const int xc0 = min(max(x0, 0), 63), xc1 = min(max(x0 + 1, 0), 63);
  const int yc0 = min(max(y0, 0), 63), yc1 = min(max(y0 + 1, 0), 63);
  const float w00 = wx0 * wy0 * (float)(vx0 && vy0);
  const float w10 = wx1 * wy0 * (float)(vx1 && vy0);
  const float w01 = wx0 * wy1 * (float)(vx0 && vy1);
  const float w11 = wx1 * wy1 * (float)(vx1 && vy1);
  const half_t* fb = featT + (long)b * 1048576 + g * 32;
  const half_t* p00 = fb + (long)(yc0 * 64 + xc0) * 256;
  const half_t* p10 = fb + (long)(yc0 * 64 + xc1) * 256;
  const half_t* p01 = fb + (long)(yc1 * 64 + xc0) * 256;
  const half_t* p11 = fb + (long)(yc1 * 64 + xc1) * 256;
  half_t* kb = kvT + (long)b * 65536 + (long)t * 256 + g * 32;
#pragma unroll
  for (int c8 = 0; c8 < 4; ++c8) {
    const v8h a00 = *(const v8h*)(p00 + c8 * 8);
    const v8h a10 = *(const v8h*)(p10 + c8 * 8);
    const v8h a01 = *(const v8h*)(p01 + c8 * 8);
    const v8h a11 = *(const v8h*)(p11 + c8 * 8);
    v8h res;
#pragma unroll
    for (int e = 0; e < 8; ++e)
      res[e] = (half_t)(w00 * (float)a00[e] + w10 * (float)a10[e] +
                        w01 * (float)a01[e] + w11 * (float)a11[e]);
    *(v8h*)(kb + c8 * 8) = res;
  }
}

// ---------------------------------------------------------------------------
// Fused deformable attention: scores = q^T k / sqrt(32), softmax over 256 kv,
// out^T = attn @ v^T.  One wave per (b, h, 16-row n-tile).  q/k spatial-major
// and v channel-major, so every WMMA fragment is k-contiguous (b128 loads /
// ds_read_b128 from LDS for the renormalized attention).
// ---------------------------------------------------------------------------
__global__ void deform_attn_kernel(const half_t* __restrict__ qT,
                                   const half_t* __restrict__ kT,
                                   const half_t* __restrict__ vch,
                                   half_t* __restrict__ aoT) {
  __shared__ float  s[16 * 256];
  __shared__ half_t ah[16 * 256];
  const int lane = threadIdx.x;
  const int hi = lane >> 4, lo = lane & 15;
  const int nt = blockIdx.x, h = blockIdx.y, b = blockIdx.z;
  const int n0 = nt * 16;
  const half_t* qb = qT  + (long)b * 1048576 + h * 32;
  const half_t* kb = kT  + (long)b * 65536   + h * 32;
  const half_t* vb = vch + (long)b * 65536   + (long)h * 32 * 256;

  // A = q^T tile (16 n-rows x K=32 head dims), channels contiguous.
  const v16h aq = frag_k1(qb + (long)(n0 + lo) * 256 + hi * 8);
  const float scale = 0.17677669529663687f;  // 1/sqrt(32)

  for (int mt = 0; mt < 16; ++mt) {
    v8f acc = {};
    const v16h bk = frag_k1(kb + (long)(mt * 16 + lo) * 256 + hi * 8);
    acc = wmma32(aq, bk, acc);
#pragma unroll
    for (int r = 0; r < 8; ++r)
      s[(r + hi * 8) * 256 + mt * 16 + lo] = acc[r] * scale;
  }
  __syncthreads();

  if (lane < 16) {
    float* row = s + lane * 256;
    float mx = -1e30f;
    for (int i = 0; i < 256; ++i) mx = fmaxf(mx, row[i]);
    float sum = 0.f;
    for (int i = 0; i < 256; ++i) { const float e = __expf(row[i] - mx); sum += e; row[i] = e; }
    const float inv = 1.f / sum;
    for (int i = 0; i < 256; ++i) ah[lane * 256 + i] = (half_t)(row[i] * inv);
  }
  __syncthreads();

  half_t* ob = aoT + (long)b * 1048576 + h * 32;
  for (int dt = 0; dt < 2; ++dt) {
    v8f acc = {};
    for (int k0 = 0; k0 < 256; k0 += 32) {
      const v16h aa = frag_k1(ah + lo * 256 + k0 + hi * 8);
      const v16h bv = frag_k1(vb + (long)(dt * 16 + lo) * 256 + k0 + hi * 8);
      acc = wmma32(aa, bv, acc);
    }
#pragma unroll
    for (int r = 0; r < 8; ++r)
      ob[(long)(n0 + r + hi * 8) * 256 + dt * 16 + lo] = (half_t)acc[r];
  }
}

// ---------------------------------------------------------------------------
// Softmax over spatial axis (n=4096) per (b, channel) row, in place on f16;
// keys are channel-major so rows are contiguous.
// ---------------------------------------------------------------------------
__global__ void ksoftmax_kernel(half_t* __restrict__ keys) {
  __shared__ float red[256];
  half_t* row = keys + (long)blockIdx.x * 4096;
  const int t = threadIdx.x;
  float mx = -1e30f;
  for (int i = t; i < 4096; i += 256) mx = fmaxf(mx, (float)row[i]);
  red[t] = mx; __syncthreads();
  for (int sh = 128; sh > 0; sh >>= 1) { if (t < sh) red[t] = fmaxf(red[t], red[t + sh]); __syncthreads(); }
  mx = red[0]; __syncthreads();
  float sum = 0.f;
  for (int i = t; i < 4096; i += 256) sum += __expf((float)row[i] - mx);
  red[t] = sum; __syncthreads();
  for (int sh = 128; sh > 0; sh >>= 1) { if (t < sh) red[t] += red[t + sh]; __syncthreads(); }
  const float inv = 1.f / red[0];
  for (int i = t; i < 4096; i += 256) row[i] = (half_t)(__expf((float)row[i] - mx) * inv);
}

// ---------------------------------------------------------------------------
// Softmax over per-head channel axis (32) per (b, n); queries are spatial-
// major so the 32 channels are contiguous per position.
// ---------------------------------------------------------------------------
__global__ void qsoftmax_kernel(half_t* __restrict__ q) {
  const int gid = blockIdx.x * 256 + threadIdx.x;  // over B*4096 = 32768
  half_t* base = q + (long)gid * 256;
  for (int h = 0; h < 8; ++h) {
    half_t* p = base + h * 32;
    float mx = -1e30f;
#pragma unroll
    for (int d = 0; d < 32; ++d) mx = fmaxf(mx, (float)p[d]);
    float e[32], sum = 0.f;
#pragma unroll
    for (int d = 0; d < 32; ++d) { e[d] = __expf((float)p[d] - mx); sum += e[d]; }
    const float inv = 1.f / sum;
#pragma unroll
    for (int d = 0; d < 32; ++d) p[d] = (half_t)(e[d] * inv);
  }
}

// ---------------------------------------------------------------------------
// Host-side orchestration
// ---------------------------------------------------------------------------
extern "C" void kernel_launch(void* const* d_in, const int* in_sizes, int n_in,
                              void* d_out, int out_size, void* d_ws, size_t ws_size,
                              hipStream_t stream) {
  (void)in_sizes; (void)n_in; (void)out_size; (void)ws_size;
  const float* x      = (const float*)d_in[0];
  const float* conv_w = (const float*)d_in[1];
  const float* conv_b = (const float*)d_in[2];
  const float* q_w    = (const float*)d_in[3];
  const float* k_w    = (const float*)d_in[4];
  const float* v_w    = (const float*)d_in[5];
  const float* o_w    = (const float*)d_in[6];
  const float* o_b    = (const float*)d_in[7];
  const float* off_dw = (const float*)d_in[8];
  const float* off_pw = (const float*)d_in[9];
  const float* ek_w   = (const float*)d_in[10];
  const float* ek_b   = (const float*)d_in[11];
  const float* eq_w   = (const float*)d_in[12];
  const float* eq_b   = (const float*)d_in[13];
  const float* ev_w   = (const float*)d_in[14];
  const float* ev_b   = (const float*)d_in[15];
  const float* er_w   = (const float*)d_in[16];
  const float* er_b   = (const float*)d_in[17];
  float* out = (float*)d_out;

  // ---- workspace carve ----
  char* p = (char*)d_ws;
  auto carve = [&](size_t bytes) -> char* {
    char* r = p; p += (bytes + 255) & ~(size_t)255; return r;
  };
  const long NF = 4096, B = 8, C = 256, Cin = 512, NK = 256;
  half_t* xT   = (half_t*)carve((size_t)B * Cin * NF * 2);  // (B, 4096, 512)
  half_t* wch  = (half_t*)carve((size_t)C * Cin * 2);
  half_t* wqh  = (half_t*)carve((size_t)C * C * 2);
  half_t* wkh  = (half_t*)carve((size_t)C * C * 2);
  half_t* wvh  = (half_t*)carve((size_t)C * C * 2);
  half_t* woh  = (half_t*)carve((size_t)C * C * 2);
  half_t* wekh = (half_t*)carve((size_t)C * C * 2);
  half_t* weqh = (half_t*)carve((size_t)C * C * 2);
  half_t* wevh = (half_t*)carve((size_t)C * C * 2);
  half_t* werh = (half_t*)carve((size_t)C * C * 2);
  half_t* featT = (half_t*)carve((size_t)B * C * NF * 2);   // (B, 4096, 256)
  half_t* qT    = (half_t*)carve((size_t)B * C * NF * 2);   // (B, 4096, 256)
  float*  px    = (float*)carve((size_t)64 * 256 * 4);
  float*  py    = (float*)carve((size_t)64 * 256 * 4);
  half_t* kvT   = (half_t*)carve((size_t)B * C * NK * 2);   // (B, 256, 256) spatial-major
  half_t* kdT   = (half_t*)carve((size_t)B * C * NK * 2);   // (B, 256, 256) spatial-major
  half_t* vch   = (half_t*)carve((size_t)B * C * NK * 2);   // (B, 256, 256) channel-major
  half_t* aoT   = (half_t*)carve((size_t)B * C * NF * 2);   // (B, 4096, 256)
  half_t* ekh   = (half_t*)carve((size_t)B * C * NF * 2);   // channel-major
  half_t* eqT   = (half_t*)carve((size_t)B * C * NF * 2);   // spatial-major
  half_t* evh   = (half_t*)carve((size_t)B * C * NF * 2);   // channel-major
  half_t* ctxh  = (half_t*)carve((size_t)64 * 32 * 32 * 2); // (z, k, v)
  half_t* attT  = (half_t*)carve((size_t)B * C * NF * 2);   // (B, 4096, 256)

  const long ZB = 512L * NF;     // d_out per-batch stride
  const long SB = (long)C * NF;  // 1048576: spatial-major per-batch stride

  // ---- input transpose + weight conversions ----
  transpose_cvt_kernel<<<dim3(128, 16, 8), 256, 0, stream>>>(x, xT, (int)Cin, (int)NF);
  cvt_f32_f16_kernel<<<64, 256, 0, stream>>>(conv_w, wch, C * Cin);
  cvt_f32_f16_kernel<<<32, 256, 0, stream>>>(q_w,  wqh,  C * C);
  cvt_f32_f16_kernel<<<32, 256, 0, stream>>>(k_w,  wkh,  C * C);
  cvt_f32_f16_kernel<<<32, 256, 0, stream>>>(v_w,  wvh,  C * C);
  cvt_f32_f16_kernel<<<32, 256, 0, stream>>>(o_w,  woh,  C * C);
  cvt_f32_f16_kernel<<<32, 256, 0, stream>>>(ek_w, wekh, C * C);
  cvt_f32_f16_kernel<<<32, 256, 0, stream>>>(eq_w, weqh, C * C);
  cvt_f32_f16_kernel<<<32, 256, 0, stream>>>(ev_w, wevh, C * C);
  cvt_f32_f16_kernel<<<32, 256, 0, stream>>>(er_w, werh, C * C);

  // ---- shared stem: featT = (conv_w @ x + conv_b)^T ----
  gemm_nt_kernel<4, false><<<dim3(32, 4, 8), 256, 0, stream>>>(
      wch, 0, 0, (int)Cin, xT, (long)Cin * NF, 0, (int)Cin,
      conv_b, nullptr, 0, 0,
      featT, SB, 0, (int)C, nullptr, 0, 0, nullptr, 0, 0,
      (int)NF, 0, (int)C, (int)NF, (int)Cin);

  // ---- deformable branch ----
  gemm_nt_kernel<4, false><<<dim3(32, 4, 8), 256, 0, stream>>>(
      wqh, 0, 0, (int)C, featT, SB, 0, (int)C,
      nullptr, nullptr, 0, 0,
      qT, SB, 0, (int)C, nullptr, 0, 0, nullptr, 0, 0,
      (int)NF, 0, (int)C, (int)NF, (int)C);
  offsets_kernel<<<64, 256, 0, stream>>>(qT, off_dw, off_pw, px, py);
  sample_kernel<<<64, 256, 0, stream>>>(featT, px, py, kvT);
  gemm_nt_kernel<4, false><<<dim3(2, 4, 8), 256, 0, stream>>>(
      wkh, 0, 0, (int)C, kvT, (long)C * NK, 0, (int)C,
      nullptr, nullptr, 0, 0,
      kdT, (long)C * NK, 0, (int)C, nullptr, 0, 0, nullptr, 0, 0,
      (int)NK, 0, (int)C, (int)NK, (int)C);
  gemm_nt_kernel<4, false><<<dim3(2, 4, 8), 256, 0, stream>>>(
      wvh, 0, 0, (int)C, kvT, (long)C * NK, 0, (int)C,
      nullptr, nullptr, 0, 0,
      nullptr, 0, 0, (int)C, vch, (long)C * NK, 0, nullptr, 0, 0,
      (int)NK, 0, (int)C, (int)NK, (int)C);
  deform_attn_kernel<<<dim3(256, 8, 8), 32, 0, stream>>>(qT, kdT, vch, aoT);
  // o-projection straight into d_out channels [0,256), channel-major f32
  gemm_nt_kernel<4, false><<<dim3(32, 4, 8), 256, 0, stream>>>(
      woh, 0, 0, (int)C, aoT, SB, 0, (int)C,
      o_b, nullptr, 0, 0,
      nullptr, 0, 0, (int)C, nullptr, 0, 0, out, ZB, 0,
      (int)NF, 0, (int)C, (int)NF, (int)C);

  // ---- efficient branch ----
  gemm_nt_kernel<4, false><<<dim3(32, 4, 8), 256, 0, stream>>>(
      wekh, 0, 0, (int)C, featT, SB, 0, (int)C,
      ek_b, nullptr, 0, 0,
      nullptr, 0, 0, (int)C, ekh, SB, 0, nullptr, 0, 0,
      (int)NF, 0, (int)C, (int)NF, (int)C);
  gemm_nt_kernel<4, false><<<dim3(32, 4, 8), 256, 0, stream>>>(
      weqh, 0, 0, (int)C, featT, SB, 0, (int)C,
      eq_b, nullptr, 0, 0,
      eqT, SB, 0, (int)C, nullptr, 0, 0, nullptr, 0, 0,
      (int)NF, 0, (int)C, (int)NF, (int)C);
  gemm_nt_kernel<4, false><<<dim3(32, 4, 8), 256, 0, stream>>>(
      wevh, 0, 0, (int)C, featT, SB, 0, (int)C,
      ev_b, nullptr, 0, 0,
      nullptr, 0, 0, (int)C, evh, SB, 0, nullptr, 0, 0,
      (int)NF, 0, (int)C, (int)NF, (int)C);
  ksoftmax_kernel<<<2048, 256, 0, stream>>>(ekh);  // over n per (b, channel)
  qsoftmax_kernel<<<128, 256, 0, stream>>>(eqT);   // over dh per (b, n)

  // context[z=(b,h)] (32x32 k-major) = key_sm (32x4096) @ values^T (4096x32)
  // A rows and B^T rows are both channel-major rows -> fully vectorized.
  gemm_nt_kernel<2, false><<<dim3(1, 1, 64), 256, 0, stream>>>(
      ekh, 32L * NF, 0, (int)NF, evh, 32L * NF, 0, (int)NF,
      nullptr, nullptr, 0, 0,
      nullptr, 0, 0, 32, ctxh, 1024, 0, nullptr, 0, 0,
      32, 0, 32, 32, (int)NF);
  // attended^T[z] = (ctx^T (32x32) @ query_sm (32x4096))^T, spatial-major out;
  // z=(b,h) views need the split batch offsets (zlog=3).
  gemm_nt_kernel<2, true><<<dim3(32, 1, 64), 256, 0, stream>>>(
      ctxh, 8192, 1024, 32, eqT, SB, 32, (int)C,
      nullptr, nullptr, 0, 0,
      attT, SB, 32, (int)C, nullptr, 0, 0, nullptr, 0, 0,
      (int)NF, 3, 32, (int)NF, 32);
  // r-projection + bias + residual(featT) into d_out channels [256,512)
  gemm_nt_kernel<4, false><<<dim3(32, 4, 8), 256, 0, stream>>>(
      werh, 0, 0, (int)C, attT, SB, 0, (int)C,
      er_b, featT, SB, 0,
      nullptr, 0, 0, (int)C, nullptr, 0, 0, out + 256L * NF, ZB, 0,
      (int)NF, 0, (int)C, (int)NF, (int)C);
}
